// SparseIM_29892972380504
// MI455X (gfx1250) — matrix operand order repaired and checked by reading, not compiled, not run on hardware
//
#include <hip/hip_runtime.h>
#include <hip/hip_bf16.h>
#include <math.h>

// ---------------------------------------------------------------------------
// Problem constants (reference: N=4096, L=32, E=512, M=512)
// ---------------------------------------------------------------------------
#define N_NODES 4096
#define L_EDGES 32
#define E_DIM   512
#define M_DIM   512
#define K_DIM   1024                       // E + M
#define R_TOTAL (N_NODES * L_EDGES)

#define NODES_PER_BLOCK 2
#define ROWS    (NODES_PER_BLOCK * L_EDGES)   // 64 rows per workgroup

#define A_STRIDE  1032                     // bf16 halves per A row (1024 + 8 pad)
#define FM_STRIDE 516                      // f32 per final_mem row (512 + 4 pad)
#define LDS_A_BYTES   (ROWS * A_STRIDE * 2)       // 132096 (== 64*516*4, reused)
#define LDS_RED_OFF   LDS_A_BYTES                  // 512 f32 reduction scratch
#define LDS_ROW_OFF   (LDS_A_BYTES + 2048)         // 256 f32 per-row scratch
#define LDS_TOTAL     (LDS_A_BYTES + 2048 + 1024)  // 135168 bytes

// HardConcrete constants
#define HC_LOC_BIAS   3.0f
#define HC_BETA_GZ    (-0.536479304144700125f)     // BETA*log(0.2)
#define HC_PEN_SCALE  (1.0f / (131072.0f * 32.0f))

typedef __attribute__((ext_vector_type(16))) __bf16       v16bf;
typedef __attribute__((ext_vector_type(8)))  float        v8f;
typedef __attribute__((ext_vector_type(4)))  unsigned int v4u;

union BF16x16 { v4u u[2]; v16bf b; };

__device__ __forceinline__ unsigned short f32_to_bf16_rne(float f) {
  unsigned int x = __builtin_bit_cast(unsigned int, f);
  x += 0x7FFFu + ((x >> 16) & 1u);          // round-to-nearest-even
  return (unsigned short)(x >> 16);
}

// B fragment per ISA layout: halves [o..o+7] then [o+16..o+23] of its column
__device__ __forceinline__ v16bf load_b_frag(const unsigned short* p) {
  BF16x16 t;
  t.u[0] = *(const v4u*)(p);
  t.u[1] = *(const v4u*)(p + 16);
  return t.b;
}

// A fragment per ISA layout: halves [o..o+7] then [o+16..o+23]
__device__ __forceinline__ v16bf load_a_frag(const unsigned short* p) {
  BF16x16 t;
  t.u[0] = *(const v4u*)(p);
  t.u[1] = *(const v4u*)(p + 16);
  return t.b;
}

// ---------------------------------------------------------------------------
// Kernel 0: init the three scalar outputs (atomics accumulate into [0],[1])
// ---------------------------------------------------------------------------
__global__ void sparseim_init_kernel(float* __restrict__ out) {
  out[0] = 0.0f;                       // penalty_total
  out[1] = 0.0f;                       // remain_edge_batch
  out[2] = (float)R_TOTAL;             // total_edge_batch
}

// ---------------------------------------------------------------------------
// Kernel 1: W0[k=1024][m=512] f32  ->  Wt[m=512][k=1024] bf16 (workspace)
// ---------------------------------------------------------------------------
__global__ void sparseim_cvtw0_kernel(const float* __restrict__ W0,
                                      unsigned short* __restrict__ Wt) {
  int id = blockIdx.x * 256 + threadIdx.x;     // 512*1024 elements
  int mo = id >> 10;
  int k  = id & 1023;
  Wt[id] = f32_to_bf16_rne(W0[k * M_DIM + mo]);
}

// ---------------------------------------------------------------------------
// Kernel 2: fused pipeline. One workgroup (8 waves) per 2 nodes (64 rows).
// ---------------------------------------------------------------------------
__global__ void __launch_bounds__(256, 1)
sparseim_main_kernel(const float* __restrict__ feat,
                     const float* __restrict__ ts,
                     const float* __restrict__ last_update,
                     const float* __restrict__ memory,
                     const float* __restrict__ u_in,
                     const float* __restrict__ bfreq,
                     const float* __restrict__ phase,
                     const float* __restrict__ b0,
                     const float* __restrict__ ln_g,
                     const float* __restrict__ ln_b,
                     const float* __restrict__ W1,
                     const float* __restrict__ b1,
                     const int*   __restrict__ eids,
                     const unsigned short* __restrict__ Wt,
                     float* __restrict__ out) {
  extern __shared__ __align__(16) unsigned char smem[];
  unsigned short* sA   = (unsigned short*)smem;          // bf16 A panel
  float*          sFM  = (float*)smem;                   // f32 final_mem (alias)
  float*          sRed = (float*)(smem + LDS_RED_OFF);   // 512 f32
  float*          sRow = (float*)(smem + LDS_ROW_OFF);   // 256 f32

  const int tid   = threadIdx.x;
  const int n0    = blockIdx.x * NODES_PER_BLOCK;
  const int wave  = tid >> 5;
  const int lane  = tid & 31;
  const int lhalf = lane >> 4;      // 0: lanes 0-15, 1: lanes 16-31
  const int lmod  = lane & 15;

  float* outScalars = out;
  float* outSflat   = out + 3;
  float* outMem     = out + 3 + R_TOTAL;

  // --- Per-row dt (hoisted so A-build doesn't reload ts per k) ------------
  if (tid < ROWS) {
    const int node = n0 + (tid >> 5);
    sRow[tid] = ts[node * L_EDGES + (tid & 31)] - last_update[node];
  }
  __syncthreads();

  // --- Stage A panel: msgs[r, k] in bf16 (fused time-encode + concat) -----
  for (int idx = tid; idx < ROWS * K_DIM; idx += 256) {
    const int r = idx >> 10;               // row 0..63
    const int k = idx & 1023;
    const int node = n0 + (r >> 5);
    const int l    = r & 31;
    float v;
    if (k < E_DIM) {
      v = feat[((size_t)node * L_EDGES + l) * E_DIM + k] +
          __cosf(sRow[r] * bfreq[k] + phase[k]);
    } else {
      v = memory[(size_t)node * M_DIM + (k - E_DIM)];
    }
    sA[r * A_STRIDE + k] = f32_to_bf16_rne(v);
  }
  __syncthreads();

  // --- GEMM: 64 rows x 512 cols, K=1024, bf16 WMMA f32-accumulate ---------
  // wave owns cols [wave*64, wave*64+63]: 4 col-tiles x 4 row-tiles
  v8f acc[4][4] = {};

  const unsigned short* aBase = sA + lmod * A_STRIDE + lhalf * 8;
  const unsigned short* bBase =
      Wt + ((size_t)(wave * 64 + lmod)) * K_DIM + lhalf * 8;

  for (int kk = 0; kk < K_DIM / 32; ++kk) {
    const int k0 = kk * 32;
    v16bf a[4];
#pragma unroll
    for (int rt = 0; rt < 4; ++rt)
      a[rt] = load_a_frag(aBase + rt * 16 * A_STRIDE + k0);
#pragma unroll
    for (int ct = 0; ct < 4; ++ct) {
      const v16bf b = load_b_frag(bBase + (size_t)ct * 16 * K_DIM + k0);
#pragma unroll
      for (int rt = 0; rt < 4; ++rt)
        acc[rt][ct] = __builtin_amdgcn_wmma_f32_16x16x32_bf16(
            false, a[rt], false, b, (short)0, acc[rt][ct], false, false);
    }
  }
  __syncthreads();   // all A reads done -> safe to alias LDS as final_mem

  // --- Spill final_mem (+ b0) to LDS per documented C/D layout ------------
#pragma unroll
  for (int ct = 0; ct < 4; ++ct) {
    const int col = wave * 64 + ct * 16 + lmod;
    const float b0v = b0[col];
#pragma unroll
    for (int rt = 0; rt < 4; ++rt) {
#pragma unroll
      for (int j = 0; j < 8; ++j) {
        const int row = rt * 16 + lhalf * 8 + j;
        sFM[row * FM_STRIDE + col] = acc[rt][ct][j] + b0v;
      }
    }
  }
  __syncthreads();

  // --- mem_out = mean over L for both nodes -------------------------------
  for (int cc = tid; cc < NODES_PER_BLOCK * M_DIM; cc += 256) {
    const int nd = cc >> 9;           // node within block
    const int c  = cc & 511;
    float s = 0.0f;
#pragma unroll 4
    for (int l = 0; l < L_EDGES; ++l) s += sFM[(nd * L_EDGES + l) * FM_STRIDE + c];
    outMem[(size_t)(n0 + nd) * M_DIM + c] = s * (1.0f / (float)L_EDGES);
  }

  // --- LayerNorm stats: 4 threads per row ---------------------------------
  const int row = tid >> 2;
  const int sub = tid & 3;
  float sum = 0.0f, sq = 0.0f;
  for (int c = sub; c < M_DIM; c += 4) {
    const float v = sFM[row * FM_STRIDE + c];
    sum += v; sq += v * v;
  }
  sRed[row * 4 + sub] = sum;
  sRed[256 + row * 4 + sub] = sq;
  __syncthreads();
  if (tid < ROWS) {
    float s = 0.0f, q = 0.0f;
#pragma unroll
    for (int i = 0; i < 4; ++i) { s += sRed[tid * 4 + i]; q += sRed[256 + tid * 4 + i]; }
    const float mu  = s * (1.0f / (float)M_DIM);
    const float var = q * (1.0f / (float)M_DIM) - mu * mu;
    sRow[tid]        = mu;                 // dt no longer needed
    sRow[ROWS + tid] = rsqrtf(var + 1e-5f);
  }
  __syncthreads();

  // --- gate = <layernorm(final_mem), W1> + b1 -----------------------------
  const float mu  = sRow[row];
  const float inv = sRow[ROWS + row];
  float gp = 0.0f;
  for (int c = sub; c < M_DIM; c += 4) {
    const float h = (sFM[row * FM_STRIDE + c] - mu) * inv * ln_g[c] + ln_b[c];
    gp += h * W1[c];
  }
  sRed[row * 4 + sub] = gp;
  __syncthreads();

  // --- HardConcrete gate + scatter (one thread per row) -------------------
  if (tid < ROWS) {
    float g = 0.0f;
#pragma unroll
    for (int i = 0; i < 4; ++i) g += sRed[tid * 4 + i];
    const float gate = g + b1[0];
    const float x = gate + HC_LOC_BIAS;
    const int node = n0 + (tid >> 5);
    const int l    = tid & 31;
    const float uu = u_in[(size_t)node * L_EDGES + l];
    const float logit = logf(uu) - log1pf(-uu);
    const float s   = 1.0f / (1.0f + expf(-((logit + x) * 3.0f)));   // /BETA
    const float pen = 1.0f / (1.0f + expf(-(x - HC_BETA_GZ)));
    float sc = s * 1.2f - 0.2f;                                      // zeta-gamma, gamma
    sc = fminf(fmaxf(sc, 0.0f), 1.0f);
    const float hard = (sc > 0.5f) ? 1.0f : 0.0f;                    // straight-through fwd
    outSflat[eids[(size_t)node * L_EDGES + l]] = hard;               // eids is a permutation
    sRow[2 * ROWS + tid] = pen;
    sRow[3 * ROWS + tid] = hard;
  }
  __syncthreads();
  if (tid == 0) {
    float ps = 0.0f, rs = 0.0f;
#pragma unroll
    for (int i = 0; i < ROWS; ++i) {
      ps += sRow[2 * ROWS + i];
      rs += sRow[3 * ROWS + i];
    }
    atomicAdd(&outScalars[0], ps * HC_PEN_SCALE);   // penalty/(N*L*L)
    atomicAdd(&outScalars[1], rs);                  // sum of 0/1 -> exact
  }
}

// ---------------------------------------------------------------------------
extern "C" void kernel_launch(void* const* d_in, const int* in_sizes, int n_in,
                              void* d_out, int out_size, void* d_ws, size_t ws_size,
                              hipStream_t stream) {
  (void)in_sizes; (void)n_in; (void)out_size; (void)ws_size;
  const float* feat        = (const float*)d_in[0];
  const float* ts          = (const float*)d_in[1];
  const float* last_update = (const float*)d_in[2];
  const float* memory      = (const float*)d_in[3];
  const float* u_in        = (const float*)d_in[4];
  const float* bfreq       = (const float*)d_in[5];
  const float* phase       = (const float*)d_in[6];
  const float* W0          = (const float*)d_in[7];
  const float* b0          = (const float*)d_in[8];
  const float* ln_g        = (const float*)d_in[9];
  const float* ln_b        = (const float*)d_in[10];
  const float* W1          = (const float*)d_in[11];
  const float* b1          = (const float*)d_in[12];
  const int*   eids        = (const int*)d_in[13];

  float* out = (float*)d_out;
  unsigned short* Wt = (unsigned short*)d_ws;   // 512*1024 bf16 = 1 MB

  sparseim_init_kernel<<<1, 1, 0, stream>>>(out);
  sparseim_cvtw0_kernel<<<(K_DIM * M_DIM) / 256, 256, 0, stream>>>(W0, Wt);
  sparseim_main_kernel<<<N_NODES / NODES_PER_BLOCK, 256, LDS_TOTAL, stream>>>(
      feat, ts, last_update, memory, u_in, bfreq, phase,
      b0, ln_g, ln_b, W1, b1, eids, Wt, out);
}
